// FeedForward_50268297232960
// MI455X (gfx1250) — 8-bit matrix operand order repaired and checked by reading, compiled, not run on hardware
//
#include <hip/hip_runtime.h>
#include <cstdint>
#include <cstddef>


typedef int          v8i  __attribute__((ext_vector_type(8)));
typedef int          v4i  __attribute__((ext_vector_type(4)));
typedef unsigned int v4u  __attribute__((ext_vector_type(4)));

#define EPS_F 1e-5f

static constexpr int NTOK = 8192;
static constexpr int DDIM = 4096;
static constexpr int HDIM = 11008;

static constexpr int TILE_M = 32;
static constexpr int TILE_N = 256;
static constexpr int LDK    = 128;

__device__ __forceinline__ void tdm_load_2d(unsigned lds_off, const void* gptr,
                                            unsigned tileK, unsigned tileRows,
                                            unsigned long long rowStride) {
  unsigned long long ga = (unsigned long long)gptr;
  v4u g0;
  g0[0] = 1u;
  g0[1] = lds_off;
  g0[2] = (unsigned)(ga & 0xffffffffu);
  g0[3] = (unsigned)((ga >> 32) & 0x01ffffffu)
        | (2u << 30);
  unsigned td0 = (unsigned)rowStride;
  unsigned td1 = 0x7fffffffu;
  unsigned long long s0 = rowStride;
  v8i g1;
  g1[0] = 0;
  g1[1] = (int)((td0 & 0xffffu) << 16);
  g1[2] = (int)(((td0 >> 16) & 0xffffu) | ((td1 & 0xffffu) << 16));
  g1[3] = (int)(((td1 >> 16) & 0xffffu) | ((tileK & 0xffffu) << 16));
  g1[4] = (int)(tileRows & 0xffffu);
  g1[5] = (int)(s0 & 0xffffffffu);
  g1[6] = (int)((s0 >> 32) & 0xffffu);
  g1[7] = 0;
  v4i gz4 = {0, 0, 0, 0};
  v8i gz8 = {0, 0, 0, 0, 0, 0, 0, 0};
  __builtin_amdgcn_tensor_load_to_lds(g0, g1, gz4, gz4, gz8, 0);
}

__device__ __forceinline__ v8i load_a_frag(const int8_t* lds, int m0w, int lane,
                                           int ksub) {
  int m = m0w + (lane & 15);
  const int* p = (const int*)(lds + m * LDK + ksub + ((lane & 16) ? 8 : 0));
  v8i a;
  a[0] = p[0];  a[1] = p[1];
  a[2] = p[4];  a[3] = p[5];
  a[4] = p[8];  a[5] = p[9];
  a[6] = p[12]; a[7] = p[13];
  return a;
}

__device__ __forceinline__ v8i load_b_frag(const int8_t* lds, int colBase, int lane,
                                           int ksub) {
  int col = colBase + (lane & 15);
  const int2* p = (const int2*)(lds + col * LDK + ksub);
  int q = (lane & 16) ? 1 : 0;
  int2 c0 = p[q];
  int2 c1 = p[2 + q];
  int2 c2 = p[4 + q];
  int2 c3 = p[6 + q];
  v8i b;
  b[0] = c0.x; b[1] = c0.y; b[2] = c1.x; b[3] = c1.y;
  b[4] = c2.x; b[5] = c2.y; b[6] = c3.x; b[7] = c3.y;
  return b;
}

__global__ __launch_bounds__(256) void abssum_kernel(const float* __restrict__ w,
                                                     size_t n, float* __restrict__ out) {
  __shared__ float sh[256];
  float s = 0.0f;
  for (size_t i = (size_t)blockIdx.x * 256u + threadIdx.x; i < n;
       i += (size_t)gridDim.x * 256u)
    s += fabsf(w[i]);
  sh[threadIdx.x] = s;
  __syncthreads();
  for (int o = 128; o > 0; o >>= 1) {
    if ((int)threadIdx.x < o) sh[threadIdx.x] += sh[threadIdx.x + o];
    __syncthreads();
  }
  if (threadIdx.x == 0) atomicAdd(out, sh[0]);
}

__global__ __launch_bounds__(256) void wquant_kernel(const float* __restrict__ w,
                                                     size_t n, const float* __restrict__ sum,
                                                     float inv_count,
                                                     int8_t* __restrict__ q) {
  float mean  = fmaxf(*sum * inv_count, EPS_F);
  float scale = 1.0f / mean;
  for (size_t i = (size_t)blockIdx.x * 256u + threadIdx.x; i < n;
       i += (size_t)gridDim.x * 256u) {
    float v = rintf(w[i] * scale);
    v = fminf(fmaxf(v, -1.0f), 1.0f);
    q[i] = (int8_t)v;
  }
}

__global__ __launch_bounds__(256) void act_quant_kernel(const float* __restrict__ x,
                                                        int8_t* __restrict__ q,
                                                        float* __restrict__ dq, int len) {
  __shared__ float s_ss[256];
  __shared__ float s_mx[256];
  const int tid = threadIdx.x;
  const size_t base = (size_t)blockIdx.x * (size_t)len;
  const float* row = x + base;

  float ss = 0.0f, mx = 0.0f;
  for (int i = tid; i < len; i += 256) {
    float v = row[i];
    ss += v * v;
    mx = fmaxf(mx, fabsf(v));
  }
  s_ss[tid] = ss;
  s_mx[tid] = mx;
  __syncthreads();
  for (int o = 128; o > 0; o >>= 1) {
    if (tid < o) {
      s_ss[tid] += s_ss[tid + o];
      s_mx[tid] = fmaxf(s_mx[tid], s_mx[tid + o]);
    }
    __syncthreads();
  }
  float r     = rsqrtf(s_ss[0] / (float)len + EPS_F);
  float amax  = fmaxf(s_mx[0] * r, EPS_F);
  float scale = 127.0f / amax;
  for (int i = tid; i < len; i += 256) {
    float v = rintf(row[i] * r * scale);
    v = fminf(fmaxf(v, -128.0f), 127.0f);
    q[base + i] = (int8_t)v;
  }
  if (tid == 0) dq[blockIdx.x] = amax / 127.0f;
}

__global__ __launch_bounds__(256) void gemm_gateup_kernel(
    const int8_t* __restrict__ Aq, const int8_t* __restrict__ Bg,
    const int8_t* __restrict__ Bu, const float* __restrict__ dqx,
    const float* __restrict__ sums, float* __restrict__ Hout) {
  __shared__ int8_t ldsA[2][TILE_M * LDK];
  __shared__ int8_t ldsBg[2][TILE_N * LDK];
  __shared__ int8_t ldsBu[2][TILE_N * LDK];

  const int tid   = threadIdx.x;
  const int lane  = tid & 31;
  const int wave  = tid >> 5;
  const int m0w   = (wave & 1) * 16;
  const int n0w   = (wave >> 1) * 64;
  const int mBase = blockIdx.x * TILE_M;
  const int nBase = blockIdx.y * TILE_N;

  const int8_t* Arow = Aq + (size_t)mBase * DDIM;
  const int8_t* Bgrow = Bg + (size_t)nBase * DDIM;
  const int8_t* Burow = Bu + (size_t)nBase * DDIM;

  v8i accg[4] = {};
  v8i accu[4] = {};

  constexpr int NKT = DDIM / LDK;

  if (tid < 32) {
    tdm_load_2d((unsigned)(size_t)(void*)ldsA[0],  Arow,  LDK, TILE_M, DDIM);
    tdm_load_2d((unsigned)(size_t)(void*)ldsBg[0], Bgrow, LDK, TILE_N, DDIM);
    tdm_load_2d((unsigned)(size_t)(void*)ldsBu[0], Burow, LDK, TILE_N, DDIM);
    __builtin_amdgcn_s_wait_tensorcnt(0);
  }
  __syncthreads();

  for (int it = 0; it < NKT; ++it) {
    const int cur = it & 1;
    const int nxt = cur ^ 1;

    if (it + 1 < NKT && tid < 32) {
      int kn = (it + 1) * LDK;
      tdm_load_2d((unsigned)(size_t)(void*)ldsA[nxt],  Arow + kn,  LDK, TILE_M, DDIM);
      tdm_load_2d((unsigned)(size_t)(void*)ldsBg[nxt], Bgrow + kn, LDK, TILE_N, DDIM);
      tdm_load_2d((unsigned)(size_t)(void*)ldsBu[nxt], Burow + kn, LDK, TILE_N, DDIM);
    }

#pragma unroll
    for (int ks = 0; ks < LDK; ks += 64) {
      v8i a = load_a_frag(ldsA[cur], m0w, lane, ks);
#pragma unroll
      for (int t = 0; t < 4; ++t) {
        v8i bg = load_b_frag(ldsBg[cur], n0w + t * 16, lane, ks);
        v8i bu = load_b_frag(ldsBu[cur], n0w + t * 16, lane, ks);
        accg[t] = __builtin_amdgcn_wmma_i32_16x16x64_iu8(true, a, true, bg, accg[t],
                                                         false, false);
        accu[t] = __builtin_amdgcn_wmma_i32_16x16x64_iu8(true, a, true, bu, accu[t],
                                                         false, false);
      }
    }

    if (tid < 32) __builtin_amdgcn_s_wait_tensorcnt(0);
    __syncthreads();
  }

  const float invHD = 1.0f / ((float)HDIM * (float)DDIM);
  const float dwg = fmaxf(sums[0] * invHD, EPS_F);
  const float dwu = fmaxf(sums[1] * invHD, EPS_F);
  const int rowOff  = (lane & 16) ? 8 : 0;
  const int colLane = lane & 15;
#pragma unroll
  for (int r = 0; r < 8; ++r) {
    int row = mBase + m0w + r + rowOff;
    float dx = dqx[row];
    size_t rb = (size_t)row * HDIM;
#pragma unroll
    for (int t = 0; t < 4; ++t) {
      int col = nBase + n0w + t * 16 + colLane;
      float g = (float)accg[t][r] * dx * dwg;
      float u = (float)accu[t][r] * dx * dwu;
      float s = g / (1.0f + __expf(-g));
      Hout[rb + col] = s * u;
    }
  }
}

__global__ __launch_bounds__(256) void gemm_down_kernel(
    const int8_t* __restrict__ Aq, const int8_t* __restrict__ Bw,
    const float* __restrict__ dqh, const float* __restrict__ sumw,
    float* __restrict__ Out) {
  __shared__ int8_t ldsA[2][TILE_M * LDK];
  __shared__ int8_t ldsB[2][TILE_N * LDK];

  const int tid   = threadIdx.x;
  const int lane  = tid & 31;
  const int wave  = tid >> 5;
  const int m0w   = (wave & 1) * 16;
  const int n0w   = (wave >> 1) * 64;
  const int mBase = blockIdx.x * TILE_M;
  const int nBase = blockIdx.y * TILE_N;

  const int8_t* Arow = Aq + (size_t)mBase * HDIM;
  const int8_t* Brow = Bw + (size_t)nBase * HDIM;

  v8i acc[4] = {};

  constexpr int NKT = HDIM / LDK;

  if (tid < 32) {
    tdm_load_2d((unsigned)(size_t)(void*)ldsA[0], Arow, LDK, TILE_M, HDIM);
    tdm_load_2d((unsigned)(size_t)(void*)ldsB[0], Brow, LDK, TILE_N, HDIM);
    __builtin_amdgcn_s_wait_tensorcnt(0);
  }
  __syncthreads();

  for (int it = 0; it < NKT; ++it) {
    const int cur = it & 1;
    const int nxt = cur ^ 1;

    if (it + 1 < NKT && tid < 32) {
      int kn = (it + 1) * LDK;
      tdm_load_2d((unsigned)(size_t)(void*)ldsA[nxt], Arow + kn, LDK, TILE_M, HDIM);
      tdm_load_2d((unsigned)(size_t)(void*)ldsB[nxt], Brow + kn, LDK, TILE_N, HDIM);
    }

#pragma unroll
    for (int ks = 0; ks < LDK; ks += 64) {
      v8i a = load_a_frag(ldsA[cur], m0w, lane, ks);
#pragma unroll
      for (int t = 0; t < 4; ++t) {
        v8i b = load_b_frag(ldsB[cur], n0w + t * 16, lane, ks);
        acc[t] = __builtin_amdgcn_wmma_i32_16x16x64_iu8(true, a, true, b, acc[t],
                                                        false, false);
      }
    }

    if (tid < 32) __builtin_amdgcn_s_wait_tensorcnt(0);
    __syncthreads();
  }

  const float dwd = fmaxf(sumw[0] / ((float)HDIM * (float)DDIM), EPS_F);
  const int rowOff  = (lane & 16) ? 8 : 0;
  const int colLane = lane & 15;
#pragma unroll
  for (int r = 0; r < 8; ++r) {
    int row = mBase + m0w + r + rowOff;
    float dh = dqh[row] * dwd;
    size_t rb = (size_t)row * DDIM;
#pragma unroll
    for (int t = 0; t < 4; ++t) {
      int col = nBase + n0w + t * 16 + colLane;
      Out[rb + col] = (float)acc[t][r] * dh;
    }
  }
}

extern "C" void kernel_launch(void* const* d_in, const int* in_sizes, int n_in,
                              void* d_out, int out_size, void* d_ws, size_t ws_size,
                              hipStream_t stream) {
  (void)in_sizes; (void)n_in; (void)out_size; (void)ws_size;
  const float* x  = (const float*)d_in[0];
  const float* wg = (const float*)d_in[1];
  const float* wu = (const float*)d_in[2];
  const float* wd = (const float*)d_in[3];
  float* out = (float*)d_out;

  const size_t WDSZ = (size_t)HDIM * DDIM;

  uint8_t* ws = (uint8_t*)d_ws;
  size_t off = 0;
  int8_t* xq   = (int8_t*)(ws + off); off += (size_t)NTOK * DDIM;
  float*  dqx  = (float*)(ws + off);  off += (size_t)NTOK * sizeof(float);
  float*  dqh  = (float*)(ws + off);  off += (size_t)NTOK * sizeof(float);
  float*  sums = (float*)(ws + off);  off += 256;
  int8_t* wgq  = (int8_t*)(ws + off); off += WDSZ;
  int8_t* wuq  = (int8_t*)(ws + off); off += WDSZ;
  int8_t* wdq  = (int8_t*)(ws + off); off += WDSZ;
  float*  hbuf = (float*)(ws + off);  off += (size_t)NTOK * HDIM * sizeof(float);
  int8_t* hq = wgq;

  const float invHD = 1.0f / ((float)HDIM * (float)DDIM);

  (void)hipMemsetAsync(sums, 0, 4 * sizeof(float), stream);

  abssum_kernel<<<1024, 256, 0, stream>>>(wg, WDSZ, sums + 0);
  abssum_kernel<<<1024, 256, 0, stream>>>(wu, WDSZ, sums + 1);
  abssum_kernel<<<1024, 256, 0, stream>>>(wd, WDSZ, sums + 2);
  wquant_kernel<<<2048, 256, 0, stream>>>(wg, WDSZ, sums + 0, invHD, wgq);
  wquant_kernel<<<2048, 256, 0, stream>>>(wu, WDSZ, sums + 1, invHD, wuq);
  wquant_kernel<<<2048, 256, 0, stream>>>(wd, WDSZ, sums + 2, invHD, wdq);

  act_quant_kernel<<<NTOK, 256, 0, stream>>>(x, xq, dqx, DDIM);

  dim3 g1(NTOK / TILE_M, HDIM / TILE_N);
  gemm_gateup_kernel<<<g1, 256, 0, stream>>>(xq, wgq, wuq, dqx, sums, hbuf);

  act_quant_kernel<<<NTOK, 256, 0, stream>>>(hbuf, hq, dqh, HDIM);

  dim3 g2(NTOK / TILE_M, DDIM / TILE_N);
  gemm_down_kernel<<<g2, 256, 0, stream>>>(hq, wdq, dqh, sums + 2, out);
}
